// SelfAttention_41205916238011
// MI455X (gfx1250) — matrix operand order repaired and checked by reading, not compiled, not run on hardware
//
#include <hip/hip_runtime.h>
#include <hip/hip_bf16.h>

#define B_  2
#define L_  2048
#define C_  1024
#define H_  16
#define D_  64
#define MAX_SCALE_MUL 4.605170185988092f  /* ln(100) */

typedef __attribute__((ext_vector_type(16))) __bf16 v16bf;
typedef __attribute__((ext_vector_type(8)))  float  v8f;
typedef unsigned int v4u  __attribute__((ext_vector_type(4)));
typedef int          v8i_ __attribute__((ext_vector_type(8)));
typedef int          v4i_ __attribute__((ext_vector_type(4)));

typedef __attribute__((address_space(3))) unsigned short* lds_u16p;

union FragAB { v16bf v; unsigned short u[16]; unsigned int w[8]; };
union FragC  { v8f   v; float f[8]; };

__device__ __forceinline__ unsigned short f2bf(float f) {
  union { float f; unsigned int u; } a; a.f = f;
  unsigned int u = a.u;
  u += 0x7FFFu + ((u >> 16) & 1u);   // round-to-nearest-even
  return (unsigned short)(u >> 16);
}

__device__ __forceinline__ v8f wmma_bf16(const FragAB& a, const FragAB& b, v8f c) {
  return __builtin_amdgcn_wmma_f32_16x16x32_bf16(false, a.v, false, b.v,
                                                 (short)0, c, false, false);
}

// ---- TDM: DMA a [32 rows x 64 bf16] row-major tile (row stride 64) to LDS ---
__device__ __forceinline__ void tdm_load_k_tile(unsigned lds_byte_off,
                                                const unsigned short* gptr) {
  unsigned long long ga = (unsigned long long)gptr;
  v4u g0;
  g0[0] = 1u;                                       // count=1, user mode
  g0[1] = lds_byte_off;                             // lds_addr (bytes)
  g0[2] = (unsigned)(ga & 0xFFFFFFFFull);           // global_addr[31:0]
  g0[3] = ((unsigned)((ga >> 32) & 0x1FFFFFFull))   // global_addr[56:32]
          | (2u << 30);                             // type = 2 ("image")
  v8i_ g1;
  g1[0] = 0x00010000;      // workgroup_mask=0, data_size=1 (2 bytes)
  g1[1] = (64 << 16);      // tensor_dim0 = 64 (elements)
  g1[2] = (32 << 16);      // tensor_dim0 hi=0 | tensor_dim1 = 32
  g1[3] = (64 << 16);      // tensor_dim1 hi=0 | tile_dim0 = 64
  g1[4] = 32;              // tile_dim1 = 32, tile_dim2 = 0
  g1[5] = 64;              // tensor_dim0_stride = 64 (elements)
  g1[6] = 0;               // stride hi | tensor_dim1_stride lo = 0
  g1[7] = 0;
  v4i_ z4 = {0, 0, 0, 0};
#if defined(__clang_major__) && (__clang_major__ >= 23)
  v8i_ z8 = {0, 0, 0, 0, 0, 0, 0, 0};
  __builtin_amdgcn_tensor_load_to_lds(g0, g1, z4, z4, z8, 0);
#else
  __builtin_amdgcn_tensor_load_to_lds(g0, g1, z4, z4, 0);
#endif
}

// ---------------- conversion / setup kernels ----------------

__global__ void cvt_f32_to_bf16(const float* __restrict__ src,
                                unsigned short* __restrict__ dst, int n) {
  int i = blockIdx.x * blockDim.x + threadIdx.x;
  if (i < n) dst[i] = f2bf(src[i]);
}

__global__ void build_bias3c(const float* __restrict__ qb,
                             const float* __restrict__ vb,
                             float* __restrict__ bias3c) {
  int i = blockIdx.x * blockDim.x + threadIdx.x;
  if (i >= 3 * C_) return;
  float v = 0.f;
  if (i < C_) v = qb[i];
  else if (i >= 2 * C_) v = vb[i - 2 * C_];
  bias3c[i] = v;
}

// --------- WMMA GEMM, 32x64 tile per wave: C[M,N] = A[M,K]*W[N,K]^T + bias[N]

__global__ void gemm_bf16(const unsigned short* __restrict__ A,
                          const unsigned short* __restrict__ W,
                          const float* __restrict__ bias,
                          float* __restrict__ Cout,
                          int M, int N, int K) {
  int wave = (blockIdx.x * blockDim.x + threadIdx.x) >> 5;
  int lane = threadIdx.x & 31;
  int nTiles = N >> 6;                 // 64 columns per wave
  int m0 = (wave / nTiles) << 5;       // 32 rows per wave
  int n0 = (wave % nTiles) << 6;
  if (m0 >= M) return;

  int half = lane >> 4;                // 0: lanes 0-15, 1: lanes 16-31
  int l16  = lane & 15;
  int akb  = half * 8;                 // A K-base per half

  FragC acc[2][4];
#pragma unroll
  for (int mi = 0; mi < 2; ++mi)
#pragma unroll
    for (int f = 0; f < 4; ++f)
#pragma unroll
      for (int r = 0; r < 8; ++r) acc[mi][f].f[r] = 0.f;

  const unsigned short* Arow0 = A + (size_t)(m0 + l16) * K;
  const unsigned short* Arow1 = A + (size_t)(m0 + 16 + l16) * K;

  for (int k0 = 0; k0 < K; k0 += 32) {
    FragAB a0, a1;
#pragma unroll
    for (int j = 0; j < 4; ++j) {
      a0.w[j]     = *(const unsigned int*)(Arow0 + k0 + akb + 2 * j);
      a0.w[4 + j] = *(const unsigned int*)(Arow0 + k0 + 16 + akb + 2 * j);
      a1.w[j]     = *(const unsigned int*)(Arow1 + k0 + akb + 2 * j);
      a1.w[4 + j] = *(const unsigned int*)(Arow1 + k0 + 16 + akb + 2 * j);
    }
#pragma unroll
    for (int f = 0; f < 4; ++f) {
      const unsigned short* Wrow =
          W + (size_t)(n0 + f * 16 + l16) * K + k0 + half * 8;
      FragAB b;
#pragma unroll
      for (int j = 0; j < 8; ++j)
        b.w[j] = *(const unsigned int*)(Wrow + 2 * j + (j >= 4 ? 8 : 0));
      acc[0][f].v = wmma_bf16(a0, b, acc[0][f].v);
      acc[1][f].v = wmma_bf16(a1, b, acc[1][f].v);
    }
  }

#pragma unroll
  for (int mi = 0; mi < 2; ++mi)
#pragma unroll
    for (int f = 0; f < 4; ++f) {
      int col = n0 + f * 16 + l16;
      float bv = bias[col];
#pragma unroll
      for (int r = 0; r < 8; ++r) {
        int row = m0 + mi * 16 + r + half * 8;
        Cout[(size_t)row * N + col] = acc[mi][f].f[r] + bv;
      }
    }
}

// ---------------- L2-norm + scale + layout ----------------------------------
// qkv[B*L,3C] -> Q,K [BH,L,D] bf16, Vt [BH,D,L] bf16

__global__ void normalize_qkv(const float* __restrict__ qkv,
                              const float* __restrict__ scale_mul,
                              unsigned short* __restrict__ Q,
                              unsigned short* __restrict__ Kb,
                              unsigned short* __restrict__ Vt) {
  int idx = blockIdx.x * blockDim.x + threadIdx.x;   // (b*L + l)*H + h
  if (idx >= B_ * L_ * H_) return;
  int h  = idx % H_;
  int bl = idx / H_;
  int l  = bl % L_;
  int b  = bl / L_;

  const float* base = qkv + (size_t)bl * (3 * C_) + h * D_;
  float sq = 0.f, sk = 0.f;
#pragma unroll 8
  for (int d = 0; d < D_; ++d) {
    float q = base[d], k = base[C_ + d];
    sq += q * q; sk += k * k;
  }
  float nq = fmaxf(sqrtf(sq), 1e-12f);
  float nk = fmaxf(sqrtf(sk), 1e-12f);
  float sm = expf(fminf(scale_mul[h], MAX_SCALE_MUL));
  float qs = sm / nq, ks = 1.f / nk;

  size_t qkRow  = ((size_t)(b * H_ + h) * L_ + l) * D_;
  size_t vtBase = ((size_t)(b * H_ + h) * D_) * L_ + l;
#pragma unroll 8
  for (int d = 0; d < D_; ++d) {
    Q [qkRow + d] = f2bf(base[d] * qs);
    Kb[qkRow + d] = f2bf(base[C_ + d] * ks);
    Vt[vtBase + (size_t)d * L_] = f2bf(base[2 * C_ + d]);
  }
}

// ---------------- fused flash attention: one wave per (b,h,16 q-rows) --------
// K tiles staged into LDS by the Tensor Data Mover, double buffered.

__global__ void flash_attn(const unsigned short* __restrict__ Q,
                           const unsigned short* __restrict__ Kb,
                           const unsigned short* __restrict__ Vt,
                           const float* __restrict__ bias,   // [L,L]
                           unsigned short* __restrict__ O) { // [B,L,C] bf16
  __shared__ unsigned short Pld[8][16 * 34];
  __shared__ unsigned short Ktile[8][2][32 * 64];   // per-wave double buffer
  int wib  = threadIdx.x >> 5;
  int wave = blockIdx.x * (blockDim.x >> 5) + wib;
  if (wave >= B_ * H_ * (L_ / 16)) return;
  int lane = threadIdx.x & 31;
  int qt = wave % (L_ / 16);
  int bh = wave / (L_ / 16);
  int h  = bh % H_;
  int b  = bh / H_;
  int q0 = qt * 16;
  int half = lane >> 4, l16 = lane & 15, akb = half * 8;

  const unsigned short* Qbh = Q  + (size_t)bh * L_ * D_;
  const unsigned short* Kbh = Kb + (size_t)bh * L_ * D_;
  const unsigned short* Vbh = Vt + (size_t)bh * D_ * L_;

  unsigned kld0 = (unsigned)reinterpret_cast<unsigned long long>(
      (lds_u16p)(&Ktile[wib][0][0]));
  const unsigned kBufBytes = 32 * 64 * 2;

  // Q A-fragments for d0 = 0, 32
  FragAB qa[2];
  const unsigned short* Qrow = Qbh + (size_t)(q0 + l16) * D_;
#pragma unroll
  for (int t = 0; t < 2; ++t) {
    int d0 = t * 32;
#pragma unroll
    for (int j = 0; j < 4; ++j) {
      qa[t].w[j]     = *(const unsigned int*)(Qrow + d0 + akb + 2 * j);
      qa[t].w[4 + j] = *(const unsigned int*)(Qrow + d0 + 16 + akb + 2 * j);
    }
  }

  FragC o[4];
#pragma unroll
  for (int f = 0; f < 4; ++f)
#pragma unroll
    for (int r = 0; r < 8; ++r) o[f].f[r] = 0.f;
  float m[8], lsum[8];
#pragma unroll
  for (int r = 0; r < 8; ++r) { m[r] = -1e30f; lsum[r] = 0.f; }

  unsigned short* P = Pld[wib];

  // preload K chunk 0 via TDM
  tdm_load_k_tile(kld0, Kbh);

  for (int it = 0; it < L_ / 32; ++it) {
    int kv0 = it * 32;
    int cur = it & 1;
    if (it + 1 < L_ / 32) {
      tdm_load_k_tile(kld0 + ((it + 1) & 1) * kBufBytes,
                      Kbh + (size_t)(kv0 + 32) * D_);
      __builtin_amdgcn_s_wait_tensorcnt(1);   // chunk `it` has landed
    } else {
      __builtin_amdgcn_s_wait_tensorcnt(0);
    }
    asm volatile("" ::: "memory");
    const unsigned short* Kl = &Ktile[wib][cur][0];

    FragC s0, s1;
#pragma unroll
    for (int r = 0; r < 8; ++r) { s0.f[r] = 0.f; s1.f[r] = 0.f; }

    // scores: S[16x32] = Q[16xD] * K^T[Dx32]  (K rows from LDS)
#pragma unroll
    for (int t = 0; t < 2; ++t) {
      int d0 = t * 32;
      const unsigned short* K0 = Kl + l16 * D_ + d0 + half * 8;
      const unsigned short* K1 = Kl + (16 + l16) * D_ + d0 + half * 8;
      FragAB bk0, bk1;
#pragma unroll
      for (int j = 0; j < 8; ++j) {
        bk0.w[j] = *(const unsigned int*)(K0 + 2 * j + (j >= 4 ? 8 : 0));
        bk1.w[j] = *(const unsigned int*)(K1 + 2 * j + (j >= 4 ? 8 : 0));
      }
      s0.v = wmma_bf16(qa[t], bk0, s0.v);
      s1.v = wmma_bf16(qa[t], bk1, s1.v);
    }

    // bias + online softmax; C-layout: reg r <-> row r + 8*half, col = l16
    float nm[8];
#pragma unroll
    for (int r = 0; r < 8; ++r) {
      int row = q0 + r + half * 8;
      s0.f[r] += bias[(size_t)row * L_ + kv0 + l16];
      s1.f[r] += bias[(size_t)row * L_ + kv0 + 16 + l16];
      float v = fmaxf(s0.f[r], s1.f[r]);
#pragma unroll
      for (int off = 1; off < 16; off <<= 1)
        v = fmaxf(v, __shfl_xor(v, off, 32));
      nm[r] = fmaxf(m[r], v);
    }
#pragma unroll
    for (int r = 0; r < 8; ++r) {
      float corr = expf(m[r] - nm[r]);
      m[r] = nm[r];
      lsum[r] *= corr;
#pragma unroll
      for (int f = 0; f < 4; ++f) o[f].f[r] *= corr;
      float p0 = expf(s0.f[r] - nm[r]);
      float p1 = expf(s1.f[r] - nm[r]);
      float ps = p0 + p1;
#pragma unroll
      for (int off = 1; off < 16; off <<= 1)
        ps += __shfl_xor(ps, off, 32);
      lsum[r] += ps;
      int row = r + half * 8;
      P[row * 34 + l16]      = f2bf(p0);
      P[row * 34 + 16 + l16] = f2bf(p1);
    }
    // LDS in-order per wave; wait for stores before cross-lane A-fragment read
    asm volatile("s_wait_dscnt 0" ::: "memory");

    FragAB pa;
    {
      const unsigned short* Prow = P + l16 * 34;
#pragma unroll
      for (int j = 0; j < 4; ++j) {
        pa.w[j]     = *(const unsigned int*)(Prow + akb + 2 * j);
        pa.w[4 + j] = *(const unsigned int*)(Prow + 16 + akb + 2 * j);
      }
    }

    // O[16x64] += P[16x32] * V[32x64]
#pragma unroll
    for (int f = 0; f < 4; ++f) {
      const unsigned short* Vrow =
          Vbh + (size_t)(f * 16 + l16) * L_ + kv0 + half * 8;
      FragAB vb;
#pragma unroll
      for (int j = 0; j < 8; ++j)
        vb.w[j] = *(const unsigned int*)(Vrow + 2 * j + (j >= 4 ? 8 : 0));
      o[f].v = wmma_bf16(pa, vb, o[f].v);
    }
  }

  // epilogue: divide by softmax sum, store bf16 O in [B,L,C]
#pragma unroll
  for (int r = 0; r < 8; ++r) {
    float inv = 1.f / lsum[r];
    int row = q0 + r + half * 8;
    unsigned short* Orow = O + ((size_t)b * L_ + row) * C_ + h * D_;
#pragma unroll
    for (int f = 0; f < 4; ++f)
      Orow[f * 16 + l16] = f2bf(o[f].f[r] * inv);
  }
}

// ---------------- host-side launcher ----------------------------------------

extern "C" void kernel_launch(void* const* d_in, const int* in_sizes, int n_in,
                              void* d_out, int out_size, void* d_ws, size_t ws_size,
                              hipStream_t stream) {
  const float* x         = (const float*)d_in[0];
  const float* attn_bias = (const float*)d_in[1];
  const float* qkv_w     = (const float*)d_in[2];
  const float* q_bias    = (const float*)d_in[3];
  const float* v_bias    = (const float*)d_in[4];
  const float* scale_mul = (const float*)d_in[5];
  const float* proj_w    = (const float*)d_in[6];
  const float* proj_b    = (const float*)d_in[7];
  float* out = (float*)d_out;

  const int M = B_ * L_;           // 4096
  char* ws = (char*)d_ws;
  size_t o_xbf   = 0;                                   // M*C bf16      (8 MB)
  size_t o_wqkv  = o_xbf  + (size_t)M * C_ * 2;         // 3C*C bf16     (6 MB)
  size_t o_wproj = o_wqkv + (size_t)3 * C_ * C_ * 2;    // C*C bf16      (2 MB)
  size_t o_bias3 = o_wproj + (size_t)C_ * C_ * 2;       // 3C f32
  size_t o_qkv   = o_bias3 + (size_t)3 * C_ * 4;        // M*3C f32      (48 MB)
  size_t o_Q     = o_qkv  + (size_t)M * 3 * C_ * 4;     // BH*L*D bf16   (8 MB)
  size_t o_K     = o_Q    + (size_t)B_ * H_ * L_ * D_ * 2;
  size_t o_Vt    = o_K    + (size_t)B_ * H_ * L_ * D_ * 2;
  size_t o_O     = o_qkv;  // alias: qkv f32 dead after normalize

  unsigned short* x_bf   = (unsigned short*)(ws + o_xbf);
  unsigned short* wqkv   = (unsigned short*)(ws + o_wqkv);
  unsigned short* wproj  = (unsigned short*)(ws + o_wproj);
  float*          bias3  = (float*)(ws + o_bias3);
  float*          qkv    = (float*)(ws + o_qkv);
  unsigned short* Qb     = (unsigned short*)(ws + o_Q);
  unsigned short* Kb     = (unsigned short*)(ws + o_K);
  unsigned short* Vt     = (unsigned short*)(ws + o_Vt);
  unsigned short* Obf    = (unsigned short*)(ws + o_O);

  const int TPB = 256;
  // 1) convert operands to bf16
  int n_x = M * C_, n_wq = 3 * C_ * C_, n_wp = C_ * C_;
  cvt_f32_to_bf16<<<(n_x  + TPB - 1) / TPB, TPB, 0, stream>>>(x,      x_bf,  n_x);
  cvt_f32_to_bf16<<<(n_wq + TPB - 1) / TPB, TPB, 0, stream>>>(qkv_w,  wqkv,  n_wq);
  cvt_f32_to_bf16<<<(n_wp + TPB - 1) / TPB, TPB, 0, stream>>>(proj_w, wproj, n_wp);
  build_bias3c<<<(3 * C_ + TPB - 1) / TPB, TPB, 0, stream>>>(q_bias, v_bias, bias3);

  // 2) QKV GEMM: [M,3C] = x[M,C] * qkv_w[3C,C]^T + bias3
  {
    int waves = (M / 32) * (3 * C_ / 64);
    gemm_bf16<<<waves * 32 / TPB, TPB, 0, stream>>>(x_bf, wqkv, bias3, qkv,
                                                    M, 3 * C_, C_);
  }

  // 3) L2-norm + per-head scale + layout change
  {
    int n = B_ * L_ * H_;
    normalize_qkv<<<(n + TPB - 1) / TPB, TPB, 0, stream>>>(qkv, scale_mul,
                                                           Qb, Kb, Vt);
  }

  // 4) fused flash attention (TDM-staged K tiles)
  {
    int waves = B_ * H_ * (L_ / 16);           // 4096
    flash_attn<<<waves * 32 / TPB, TPB, 0, stream>>>(Qb, Kb, Vt, attn_bias, Obf);
  }

  // 5) output projection: out[M,C] = O[M,C] * proj_w[C,C]^T + proj_b
  {
    int waves = (M / 32) * (C_ / 64);
    gemm_bf16<<<waves * 32 / TPB, TPB, 0, stream>>>(Obf, wproj, proj_b, out,
                                                    M, C_, C_);
  }
}
